// TRNModel_54460185313708
// MI455X (gfx1250) — compile-verified, matrix operand order checked by reading, not run on hardware
//
#include <hip/hip_runtime.h>
#include <hip/hip_bf16.h>

// ---------------------------------------------------------------------------
// Fused RMSNorm + LM-head GEMM + shifted cross-entropy (linear-CE) for
// B=2, T=2048, D=2048, V=50257 on gfx1250 (MI455X).
//
//   k1: h = rmsnorm(x)*w  -> bf16 workspace (16 MB, L2-resident)
//   k2: per (128-row tile, 1024-col chunk): bf16 WMMA GEMM. A-tile staged
//       into LDS by the Tensor Data Mover (tensor_load_to_lds + TENSORcnt);
//       fp32 W converted to bf16 in-register. Online softmax partials
//       (rowmax, sumexp) + target-logit gather in the epilogue.
//   k3: per-row combine of 50 chunk partials -> nll, block reduce, atomicAdd
//   k4: loss = sum / max(count, 1)
// ---------------------------------------------------------------------------

typedef __attribute__((ext_vector_type(16))) __bf16 v16bf;
typedef __attribute__((ext_vector_type(8)))  float  v8f;

#define DIM      2048
#define TSEQ     2048
#define MROWS    4096      // B*T
#define VVOC     50257
#define NCHUNK   50        // ceil(V / 1024)
#define CHUNKC   1024
#define NBPC     8         // 128-col GEMM blocks per chunk
#define KSTEP    32
#define NEG_INF  (-__builtin_inff())

#if defined(__gfx1250__) && __has_builtin(__builtin_amdgcn_tensor_load_to_lds) && \
    __has_builtin(__builtin_amdgcn_s_wait_tensorcnt)
#define HAVE_TDM 1
#endif

__device__ __forceinline__ unsigned short f2bf(float f) {
    unsigned u = __float_as_uint(f);
    unsigned r = u + 0x7FFFu + ((u >> 16) & 1u);   // round-to-nearest-even
    return (unsigned short)(r >> 16);
}
__device__ __forceinline__ unsigned pack2(float lo, float hi) {
    return (unsigned)f2bf(lo) | ((unsigned)f2bf(hi) << 16);
}

union FragU { uint4 q[2]; v16bf v; };
__device__ __forceinline__ v16bf make_frag(uint4 lo, uint4 hi) {
    FragU u; u.q[0] = lo; u.q[1] = hi; return u.v;
}

// merge two (max, sumexp) pairs
__device__ __forceinline__ void lse_combine(float m1, float s1, float m2, float s2,
                                            float& mo, float& so) {
    if (m2 == NEG_INF)      { mo = m1; so = s1; }
    else if (m1 == NEG_INF) { mo = m2; so = s2; }
    else if (m1 >= m2)      { mo = m1; so = s1 + s2 * __expf(m2 - m1); }
    else                    { mo = m2; so = s2 + s1 * __expf(m1 - m2); }
}

#ifdef HAVE_TDM
typedef unsigned u32x4 __attribute__((ext_vector_type(4)));
typedef int      i32x4 __attribute__((ext_vector_type(4)));
typedef int      i32x8 __attribute__((ext_vector_type(8)));

// TDM: DMA a 128-row x 32-col bf16 tile (row pitch 2048 elems) from the h
// tensor into LDS, packed contiguously (64 B per row) -> matches As[r][k].
// Descriptor bitfields per CDNA5 ISA 8.3/8.4 (D# group0 + group1, 2-D tile).
__device__ __forceinline__ void tdm_load_a_tile(const unsigned short* gsrc,
                                                void* lds_dst) {
    unsigned long long ga = (unsigned long long)(size_t)gsrc;
    // Flat LDS-aperture addresses truncate to the LDS byte address (ISA 10.2).
    unsigned lds = (unsigned)(size_t)lds_dst;
    u32x4 g0;
    g0.x = 1u;                                   // count=1 (valid user D#)
    g0.y = lds;                                  // lds_addr [63:32]
    g0.z = (unsigned)ga;                         // global_addr lo
    g0.w = (unsigned)(ga >> 32) | (2u << 30);    // global_addr hi | type=2
    i32x8 g1;
    g1[0] = 0x10000;                 // data_size=1 (2 bytes); no flags
    g1[1] = (int)(2048u << 16);      // tensor_dim0 = 2048 (lo16 at [63:48])
    g1[2] = (int)(4096u << 16);      // dim0 hi=0 | tensor_dim1 lo16 = 4096
    g1[3] = (int)(32u  << 16);       // dim1 hi=0 | tile_dim0 = 32
    g1[4] = 128;                     // tile_dim1 = 128 | tile_dim2 = 0
    g1[5] = 2048;                    // tensor_dim0_stride = 2048 (lo32)
    g1[6] = 0;                       // stride0 hi | tensor_dim1_stride lo = 0
    g1[7] = 0;
    i32x4 z4 = {0, 0, 0, 0};
#if __clang_major__ >= 23
    i32x8 z8 = {0, 0, 0, 0, 0, 0, 0, 0};
    __builtin_amdgcn_tensor_load_to_lds(g0, g1, z4, z4, z8, 0);
#else
    __builtin_amdgcn_tensor_load_to_lds(g0, g1, z4, z4, 0);
#endif
}
#endif

// ---------------------------------------------------------------- kernel 1 --
__global__ __launch_bounds__(256)
void rms_kernel(const float* __restrict__ x, const float* __restrict__ w,
                unsigned short* __restrict__ h) {
    const int row = blockIdx.x;
    const int tid = threadIdx.x;
    const float* xr = x + (size_t)row * DIM;

    float4 a = ((const float4*)xr)[tid * 2];
    float4 b = ((const float4*)xr)[tid * 2 + 1];
    float ss = a.x*a.x + a.y*a.y + a.z*a.z + a.w*a.w
             + b.x*b.x + b.y*b.y + b.z*b.z + b.w*b.w;

    __shared__ float red[256];
    red[tid] = ss;
    __syncthreads();
    for (int s = 128; s > 0; s >>= 1) {
        if (tid < s) red[tid] += red[tid + s];
        __syncthreads();
    }
    const float scale = rsqrtf(red[0] * (1.0f / (float)DIM) + 1e-6f);

    float4 wa = ((const float4*)w)[tid * 2];
    float4 wb = ((const float4*)w)[tid * 2 + 1];
    uint4 o;
    o.x = pack2(a.x * scale * wa.x, a.y * scale * wa.y);
    o.y = pack2(a.z * scale * wa.z, a.w * scale * wa.w);
    o.z = pack2(b.x * scale * wb.x, b.y * scale * wb.y);
    o.w = pack2(b.z * scale * wb.z, b.w * scale * wb.w);
    ((uint4*)(h + (size_t)row * DIM))[tid] = o;
}

// ---------------------------------------------------------------- kernel 2 --
// grid = (32 m-tiles, 50 v-chunks), 256 threads (8 wave32s: 4 M x 2 N)
__global__ __launch_bounds__(256)
void lce_kernel(const unsigned short* __restrict__ h,
                const float* __restrict__ lmw,
                const int* __restrict__ labels,
                float* __restrict__ pm, float* __restrict__ ps,
                float* __restrict__ targ) {
    const int mtile = blockIdx.x;
    const int chunk = blockIdx.y;
    const int tid   = threadIdx.x;
    const int lane  = tid & 31;
    const int wid   = tid >> 5;
    const int half  = lane >> 4;     // which 16-lane half
    const int lrow  = lane & 15;
    const int wave_m = wid >> 1;     // 0..3 -> 32-row strip
    const int wave_n = wid & 1;      // 0..1 -> 64-col strip

    __shared__ unsigned short As[128 * 32];   // [m][k] bf16
    __shared__ unsigned short Bs[128 * 32];   // [n][k] bf16
    __shared__ int   tcol[128];
    __shared__ float smx[128][2];
    __shared__ float ssm[128][2];
    __shared__ float runm[128], runs[128];

    if (tid < 128) {
        int row = mtile * 128 + tid;
        int t = row & (TSEQ - 1);
        int lab = (t < TSEQ - 1) ? labels[row + 1] : -100;
        tcol[tid] = (lab >= 0 && lab < VVOC) ? lab : -1;
        runm[tid] = NEG_INF;
        runs[tid] = 0.0f;
    }
    __syncthreads();

    for (int nb = 0; nb < NBPC; ++nb) {
        const int nbase = chunk * CHUNKC + nb * 128;
        if (nbase >= VVOC) break;                    // block-uniform

        v8f acc[2][4];
        #pragma unroll
        for (int tm = 0; tm < 2; ++tm)
            #pragma unroll
            for (int tn = 0; tn < 4; ++tn)
                #pragma unroll
                for (int i = 0; i < 8; ++i) acc[tm][tn][i] = 0.0f;

        for (int k0 = 0; k0 < DIM; k0 += KSTEP) {
            // ---- stage A tile (bf16 h) into LDS ----
#ifdef HAVE_TDM
            if (wid == 0)
                tdm_load_a_tile(h + (size_t)(mtile * 128) * DIM + k0, As);
#else
            {
                const int r = tid >> 1, off = (tid & 1) * 16;
                const uint4* src =
                    (const uint4*)(h + (size_t)(mtile * 128 + r) * DIM + k0 + off);
                uint4 a0 = src[0], a1 = src[1];
                uint4* dst = (uint4*)(As + r * 32 + off);
                dst[0] = a0; dst[1] = a1;
            }
#endif
            // ---- stage B tile: fp32 W -> bf16 in-register ----
            {
                const int n = tid >> 1, off = (tid & 1) * 16;
                const int v = nbase + n;
                uint4 b0 = {0,0,0,0}, b1 = {0,0,0,0};
                if (v < VVOC) {
                    const float* base = lmw + (size_t)v * DIM + k0 + off;
                    const float4* src = (const float4*)base;
                    float4 f0 = src[0], f1 = src[1], f2 = src[2], f3 = src[3];
                    if (k0 + KSTEP < DIM)
                        __builtin_prefetch(base + KSTEP, 0, 1);  // global_prefetch
                    b0.x = pack2(f0.x, f0.y); b0.y = pack2(f0.z, f0.w);
                    b0.z = pack2(f1.x, f1.y); b0.w = pack2(f1.z, f1.w);
                    b1.x = pack2(f2.x, f2.y); b1.y = pack2(f2.z, f2.w);
                    b1.z = pack2(f3.x, f3.y); b1.w = pack2(f3.z, f3.w);
                }
                uint4* dst = (uint4*)(Bs + n * 32 + off);
                dst[0] = b0; dst[1] = b1;
            }
#ifdef HAVE_TDM
            if (wid == 0) __builtin_amdgcn_s_wait_tensorcnt(0);  // A tile landed
#endif
            __syncthreads();

            // Fragments per documented CDNA5 16-bit layouts:
            // lane<16 holds K {0..7, 16..23}; lane>=16 holds K {8..15, 24..31}
            v16bf af[2], bf_[4];
            #pragma unroll
            for (int tm = 0; tm < 2; ++tm) {
                const int ar = wave_m * 32 + tm * 16 + lrow;
                const uint4* ap = (const uint4*)(As + ar * 32);
                af[tm] = make_frag(ap[half], ap[2 + half]);
            }
            #pragma unroll
            for (int tn = 0; tn < 4; ++tn) {
                const int br = wave_n * 64 + tn * 16 + lrow;
                const uint4* bp = (const uint4*)(Bs + br * 32);
                bf_[tn] = make_frag(bp[half], bp[2 + half]);
            }
            #pragma unroll
            for (int tm = 0; tm < 2; ++tm)
                #pragma unroll
                for (int tn = 0; tn < 4; ++tn)
                    acc[tm][tn] = __builtin_amdgcn_wmma_f32_16x16x32_bf16(
                        false, af[tm], false, bf_[tn],
                        (short)0, acc[tm][tn], false, false);
            __syncthreads();
        }

        // ---- epilogue: per-row block max ----
        #pragma unroll
        for (int tm = 0; tm < 2; ++tm) {
            #pragma unroll
            for (int r = 0; r < 8; ++r) {
                float mx = NEG_INF;
                #pragma unroll
                for (int tn = 0; tn < 4; ++tn) {
                    const int col = nbase + wave_n * 64 + tn * 16 + lrow;
                    if (col < VVOC) mx = fmaxf(mx, acc[tm][tn][r]);
                }
                for (int m = 1; m < 16; m <<= 1)
                    mx = fmaxf(mx, __shfl_xor(mx, m, 32));
                if (lrow == 0)
                    smx[wave_m * 32 + tm * 16 + half * 8 + r][wave_n] = mx;
            }
        }
        __syncthreads();

        // ---- epilogue: per-row sumexp + target-logit gather ----
        #pragma unroll
        for (int tm = 0; tm < 2; ++tm) {
            #pragma unroll
            for (int r = 0; r < 8; ++r) {
                const int lr = wave_m * 32 + tm * 16 + half * 8 + r;
                const float bm = smx[lr][wave_n];
                float s = 0.0f;
                #pragma unroll
                for (int tn = 0; tn < 4; ++tn) {
                    const int col = nbase + wave_n * 64 + tn * 16 + lrow;
                    const float v = acc[tm][tn][r];
                    if (col < VVOC) {
                        s += __expf(v - bm);
                        if (col == tcol[lr]) targ[mtile * 128 + lr] = v;
                    }
                }
                for (int m = 1; m < 16; m <<= 1)
                    s += __shfl_xor(s, m, 32);
                if (lrow == 0) ssm[lr][wave_n] = s;
            }
        }
        __syncthreads();

        // ---- merge the two column-halves, then into running stats ----
        if (tid < 128) {
            float bm, bs;
            lse_combine(smx[tid][0], ssm[tid][0], smx[tid][1], ssm[tid][1], bm, bs);
            float nm, nsum;
            lse_combine(runm[tid], runs[tid], bm, bs, nm, nsum);
            runm[tid] = nm; runs[tid] = nsum;
        }
        __syncthreads();
    }

    if (tid < 128) {
        const int row = mtile * 128 + tid;
        pm[(size_t)row * NCHUNK + chunk] = runm[tid];
        ps[(size_t)row * NCHUNK + chunk] = runs[tid];
    }
}

// ---------------------------------------------------------------- kernel 3 --
__global__ __launch_bounds__(256)
void finalize_rows(const float* __restrict__ pm, const float* __restrict__ ps,
                   const float* __restrict__ targ, const int* __restrict__ labels,
                   float* __restrict__ accum) {
    const int row = blockIdx.x * 256 + threadIdx.x;
    float nll = 0.0f, val = 0.0f;
    if (row < MROWS) {
        const int t = row & (TSEQ - 1);
        const int lab = (t < TSEQ - 1) ? labels[row + 1] : -100;
        if (lab >= 0) {
            float m = NEG_INF;
            for (int c = 0; c < NCHUNK; ++c)
                m = fmaxf(m, pm[(size_t)row * NCHUNK + c]);
            float s = 0.0f;
            for (int c = 0; c < NCHUNK; ++c) {
                const float pmc = pm[(size_t)row * NCHUNK + c];
                if (pmc != NEG_INF)
                    s += ps[(size_t)row * NCHUNK + c] * __expf(pmc - m);
            }
            nll = (__logf(s) + m) - targ[row];
            val = 1.0f;
        }
    }
    __shared__ float r1[256], r2[256];
    r1[threadIdx.x] = nll; r2[threadIdx.x] = val;
    __syncthreads();
    for (int s = 128; s > 0; s >>= 1) {
        if (threadIdx.x < s) {
            r1[threadIdx.x] += r1[threadIdx.x + s];
            r2[threadIdx.x] += r2[threadIdx.x + s];
        }
        __syncthreads();
    }
    if (threadIdx.x == 0) {
        atomicAdd(&accum[0], r1[0]);
        atomicAdd(&accum[1], r2[0]);
    }
}

// ---------------------------------------------------------------- kernel 4 --
__global__ void write_out(const float* __restrict__ accum, float* __restrict__ out) {
    out[0] = accum[0] / fmaxf(accum[1], 1.0f);
}

// ---------------------------------------------------------------------------
extern "C" void kernel_launch(void* const* d_in, const int* in_sizes, int n_in,
                              void* d_out, int out_size, void* d_ws, size_t ws_size,
                              hipStream_t stream) {
    const float* x      = (const float*)d_in[0];
    const float* rms_w  = (const float*)d_in[1];
    const float* lm_w   = (const float*)d_in[2];
    const int*   labels = (const int*)d_in[3];
    float*       out    = (float*)d_out;

    unsigned short* h = (unsigned short*)d_ws;                 // 4096*2048 bf16
    float* pm    = (float*)((char*)d_ws + (size_t)MROWS * DIM * sizeof(unsigned short));
    float* ps    = pm + (size_t)MROWS * NCHUNK;
    float* targ  = ps + (size_t)MROWS * NCHUNK;
    float* accum = targ + MROWS;

    hipMemsetAsync(accum, 0, 2 * sizeof(float), stream);

    rms_kernel<<<MROWS, 256, 0, stream>>>(x, rms_w, h);

    dim3 grid(MROWS / 128, NCHUNK);
    lce_kernel<<<grid, 256, 0, stream>>>(h, lm_w, labels, pm, ps, targ);

    finalize_rows<<<MROWS / 256, 256, 0, stream>>>(pm, ps, targ, labels, accum);
    write_out<<<1, 1, 0, stream>>>(accum, out);
}
